// ConvFlow_26121991094743
// MI455X (gfx1250) — matrix operand order repaired and checked by reading, not compiled, not run on hardware
//
#include <hip/hip_runtime.h>
#include <cmath>

// ---------------- problem constants (match reference) ----------------
#define CC     192          // channels
#define BB     16           // batch
#define TT     8192         // time
#define NBINS  10
#define TBOUND 5.0f

// ---------------- tiling ----------------
#define TILE 128                 // output columns per workgroup
#define HALO 13                  // total receptive radius of the 3 dilated convs
#define W0   (TILE + 2 * HALO)   // 154 loaded columns
#define W0P  168                 // row stride (f32 elems) for hbuf/zbuf (covers GEMM tile overrun)
#define LDY  200                 // per-column stride (bf16 elems) of ybuf; 200*2B = 16B-aligned, bank-friendly

typedef __attribute__((ext_vector_type(16))) __bf16       v16bf;
typedef __attribute__((ext_vector_type(8)))  float        v8f;
typedef __attribute__((ext_vector_type(4)))  unsigned int v4u;

union BFrag { v4u u[2]; v16bf v; };

__device__ __forceinline__ float gelu_f(float v) {
  // exact GELU (approximate=False): 0.5*x*(1+erf(x/sqrt(2)))
  return 0.5f * v * (1.0f + erff(v * 0.70710678118654752f));
}
__device__ __forceinline__ float softplus_f(float v) {
  return (v > 20.0f) ? v : log1pf(expf(v));
}

// ---------------------------------------------------------------------------
// Tiled GEMM  zbuf[m, lo..lo+wid) = A[m, 0..192) * ybuf + bias[m]
//   A: row-major f32 in global (rows >= rowLimit treated as zero)
//   ybuf: bf16, column-major [col*LDY + k]  (k = input channel)
//   zbuf: f32, row-major [m*W0P + col]
// Uses v_wmma_f32_16x16x32_bf16; all control flow is wave-uniform (EXEC all 1s).
// Out-of-range rows are handled with clamped indices + 0/1 multipliers so no
// per-lane EXEC branching is generated around the WMMA bursts.
// VGPR layouts used here (D per CDNA5 ISA 7.12.2):
//   A 16x32 bf16 : lane%16 = M row; lanes<16 hold K 0..15, lanes>=16 K 16..31 (contiguous, code's own order)
//   B 32x16 bf16 : lane%16 = N col; lanes<16 hold K 0..15, lanes>=16 K 16..31 (contiguous, code's own order)
//   D 16x16 f32  : VGPR r -> M = 16*mt + r + 8*(lane>=16), N = lane%16
// ---------------------------------------------------------------------------
__device__ __forceinline__ void wmma_gemm(
    const float* __restrict__ A, int rowLimit, int mtiles,
    const float* __restrict__ bias,
    const __bf16* __restrict__ ybuf,
    float* __restrict__ zbuf,
    int lo, int wid, int wave, int lane)
{
  const int ntile  = (wid + 15) >> 4;
  const int total  = mtiles * ntile;
  const int mrowA  = lane & 15;
  const int kAddA  = (lane < 16) ? 0 : 16;
  const int kBaseB = (lane < 16) ? 0 : 16;
  const int nlane  = lane & 15;
  const int mAddD  = (lane < 16) ? 0 : 8;

  v16bf afrag[6];
  float bval[8];
  int curMt = -1;

  for (int t = wave; t < total; t += 8) {
    const int mt = t / ntile;
    const int nt = t - mt * ntile;

    if (mt != curMt) {   // (re)load + convert A fragments and bias for this M tile
      curMt = mt;
      const int   m   = mt * 16 + mrowA;
      const float okf = (m < rowLimit) ? 1.0f : 0.0f;
      const float* Ar = A + ((m < rowLimit) ? m : 0) * CC;  // clamped: always valid
#pragma unroll
      for (int ks = 0; ks < 6; ++ks) {
        v16bf a;
#pragma unroll
        for (int e = 0; e < 16; ++e) {
          const int k = ks * 32 + kAddA + e;
          a[e] = (__bf16)(Ar[k] * okf);
        }
        afrag[ks] = a;
      }
      const int mb = mt * 16 + mAddD;
#pragma unroll
      for (int r = 0; r < 8; ++r) {
        const int  mm = mb + r;
        const float f = (mm < rowLimit) ? 1.0f : 0.0f;
        bval[r] = bias[(mm < rowLimit) ? mm : 0] * f;       // clamped: always valid
      }
    }

    const int ncol = lo + nt * 16 + nlane;
    v8f acc = {0.f, 0.f, 0.f, 0.f, 0.f, 0.f, 0.f, 0.f};
#pragma unroll
    for (int ks = 0; ks < 6; ++ks) {
      BFrag bf;  // 16 contiguous bf16 K-values: two aligned 16B LDS loads
      const v4u* bp = (const v4u*)(ybuf + ncol * LDY + ks * 32 + kBaseB);
      bf.u[0] = bp[0];
      bf.u[1] = bp[1];
      acc = __builtin_amdgcn_wmma_f32_16x16x32_bf16(
          false, afrag[ks], false, bf.v, (short)0, acc, false, false);
    }

    const int m0 = mt * 16 + mAddD;
#pragma unroll
    for (int r = 0; r < 8; ++r) {
      zbuf[(m0 + r) * W0P + ncol] = acc[r] + bval[r];
    }
  }
}

// ---------------------------------------------------------------------------
// Fully fused ConvFlow kernel: one workgroup = one (batch, time-tile).
// ---------------------------------------------------------------------------
__global__ __launch_bounds__(256)
void convflow_kernel(const float* __restrict__ x,
                     const float* __restrict__ xmask,
                     const float* __restrict__ w_pre,
                     const float* __restrict__ b_pre,
                     const float* __restrict__ sep_w,
                     const float* __restrict__ sep_b,
                     const float* __restrict__ pw_w,
                     const float* __restrict__ pw_b,
                     const float* __restrict__ n1_g,
                     const float* __restrict__ n1_b,
                     const float* __restrict__ n2_g,
                     const float* __restrict__ n2_b,
                     const float* __restrict__ w_proj,
                     const float* __restrict__ b_proj,
                     float* __restrict__ out,
                     float* __restrict__ partials)
{
  extern __shared__ char smem[];
  float*  hbuf = (float*)smem;                    // CC x W0P  f32  (residual stream h)
  float*  zbuf = hbuf + CC * W0P;                 // CC x W0P  f32  (conv / GEMM outputs)
  __bf16* ybuf = (__bf16*)(zbuf + CC * W0P);      // W0P cols x LDY bf16, column-major (GEMM B input)
  float*  xsh  = (float*)(ybuf + W0P * LDY);      // x0 tile (raw)
  float*  msh  = xsh + W0P;                       // mask tile

  const int tid  = threadIdx.x;
  const int lane = tid & 31;
  const int wave = tid >> 5;
  const int tile = blockIdx.x;
  const int b    = blockIdx.y;
  const int t0   = tile * TILE;

  // ---- load x0 + mask tile (zero outside sequence == conv zero padding) ----
  for (int j = tid; j < W0; j += 256) {
    const int  gt = t0 - HALO + j;
    const bool in = (gt >= 0) && (gt < TT);
    xsh[j] = in ? x[(b * 2 + 0) * TT + gt] : 0.0f;
    msh[j] = in ? xmask[b * TT + gt] : 0.0f;
  }
  __syncthreads();

  // ---- h = (x0 * w_pre + b_pre) * mask ----
  for (int idx = tid; idx < CC * W0; idx += 256) {
    const int c = idx / W0, j = idx - c * W0;
    hbuf[c * W0P + j] = (xsh[j] * w_pre[c] + b_pre[c]) * msh[j];
  }

  // ---- 3 residual blocks (fully unrolled: widths/dilations constant-fold) ----
  int lo = 0, hi = W0;
  const int dil[3] = {1, 3, 9};
#pragma unroll
  for (int layer = 0; layer < 3; ++layer) {
    const int d   = dil[layer];
    const int clo = lo + d, chi = hi - d;
    const int wid = chi - clo;
    __syncthreads();

    // depthwise dilated conv (K=3) -> zbuf
    for (int idx = tid; idx < CC * wid; idx += 256) {
      const int    c  = idx / wid;
      const int    j  = clo + (idx - c * wid);
      const float* hr = hbuf + c * W0P;
      zbuf[c * W0P + j] = hr[j - d] * sep_w[(layer * 3 + 0) * CC + c]
                        + hr[j    ] * sep_w[(layer * 3 + 1) * CC + c]
                        + hr[j + d] * sep_w[(layer * 3 + 2) * CC + c]
                        + sep_b[layer * CC + c];
    }
    __syncthreads();

    // LayerNorm over channels + GELU -> ybuf (bf16, column-major) : one thread per column
    for (int j = clo + tid; j < chi; j += 256) {
      float m1 = 0.f, m2 = 0.f;
#pragma unroll 4
      for (int c = 0; c < CC; ++c) { const float v = zbuf[c * W0P + j]; m1 += v; m2 += v * v; }
      m1 *= (1.0f / CC);
      const float rs = rsqrtf(m2 * (1.0f / CC) - m1 * m1 + 1e-5f);
#pragma unroll 4
      for (int c = 0; c < CC; ++c) {
        const float v = (zbuf[c * W0P + j] - m1) * rs * n1_g[layer * CC + c] + n1_b[layer * CC + c];
        ybuf[j * LDY + c] = (__bf16)gelu_f(v);
      }
    }
    __syncthreads();

    // pointwise 192x192 GEMM on tensor cores -> zbuf
    wmma_gemm(pw_w + layer * CC * CC, CC, 12, pw_b + layer * CC,
              ybuf, zbuf, clo, wid, wave, lane);
    __syncthreads();

    // LayerNorm + GELU + residual add into h
    for (int j = clo + tid; j < chi; j += 256) {
      float m1 = 0.f, m2 = 0.f;
#pragma unroll 4
      for (int c = 0; c < CC; ++c) { const float v = zbuf[c * W0P + j]; m1 += v; m2 += v * v; }
      m1 *= (1.0f / CC);
      const float rs = rsqrtf(m2 * (1.0f / CC) - m1 * m1 + 1e-5f);
#pragma unroll 4
      for (int c = 0; c < CC; ++c) {
        const float v = (zbuf[c * W0P + j] - m1) * rs * n2_g[layer * CC + c] + n2_b[layer * CC + c];
        hbuf[c * W0P + j] += gelu_f(v);
      }
    }
    lo = clo; hi = chi;
  }
  __syncthreads();

  // ---- projection to 29 spline params: ybuf = bf16(h * mask), GEMM (M=32 padded) ----
  for (int idx = tid; idx < CC * TILE; idx += 256) {
    const int c = idx % CC;
    const int j = HALO + idx / CC;
    ybuf[j * LDY + c] = (__bf16)(hbuf[c * W0P + j] * msh[j]);
  }
  __syncthreads();
  wmma_gemm(w_proj, 3 * NBINS - 1, 2, b_proj, ybuf, zbuf, HALO, TILE, wave, lane);
  __syncthreads();

  // ---- rational-quadratic spline on x1 (one thread per column) ----
  float lsum = 0.0f;
  if (tid < TILE) {
    const int   j  = HALO + tid;
    const int   gt = t0 + tid;
    const float m  = msh[j];

    float p[3 * NBINS - 1];
#pragma unroll
    for (int o = 0; o < 3 * NBINS - 1; ++o) p[o] = zbuf[o * W0P + j] * m;

    const float SCALE  = 0.07216878364870322f;           // 1/sqrt(192)
    const float DCONST = logf(expf(0.999f) - 1.0f);      // log(exp(1-md)-1)

    float uw[NBINS], uh[NBINS], ddraw[NBINS + 1];
#pragma unroll
    for (int k = 0; k < NBINS; ++k) { uw[k] = p[k] * SCALE; uh[k] = p[NBINS + k] * SCALE; }
    ddraw[0] = DCONST; ddraw[NBINS] = DCONST;
#pragma unroll
    for (int k = 1; k < NBINS; ++k) ddraw[k] = p[2 * NBINS + k - 1];

    // knot positions from softmax widths/heights (mw = mh = 0.001)
    float cw[NBINS + 1], ch[NBINS + 1];
    {
      float mx = uw[0];
#pragma unroll
      for (int k = 1; k < NBINS; ++k) mx = fmaxf(mx, uw[k]);
      float ex[NBINS]; float s = 0.f;
#pragma unroll
      for (int k = 0; k < NBINS; ++k) { ex[k] = expf(uw[k] - mx); s += ex[k]; }
      const float inv = 1.0f / s;
      cw[0] = -TBOUND; float cum = 0.f;
#pragma unroll
      for (int k = 0; k < NBINS; ++k) {
        cum += 0.001f + 0.99f * ex[k] * inv;
        cw[k + 1] = -TBOUND + 2.0f * TBOUND * cum;
      }
      cw[NBINS] = TBOUND;
    }
    {
      float mx = uh[0];
#pragma unroll
      for (int k = 1; k < NBINS; ++k) mx = fmaxf(mx, uh[k]);
      float ex[NBINS]; float s = 0.f;
#pragma unroll
      for (int k = 0; k < NBINS; ++k) { ex[k] = expf(uh[k] - mx); s += ex[k]; }
      const float inv = 1.0f / s;
      ch[0] = -TBOUND; float cum = 0.f;
#pragma unroll
      for (int k = 0; k < NBINS; ++k) {
        cum += 0.001f + 0.99f * ex[k] * inv;
        ch[k + 1] = -TBOUND + 2.0f * TBOUND * cum;
      }
      ch[NBINS] = TBOUND;
    }

    const float x1v = x[(b * 2 + 1) * TT + gt];
    const bool  ins = (x1v >= -TBOUND) && (x1v <= TBOUND);
    const float xi  = fminf(fmaxf(x1v, -TBOUND), TBOUND);

    int idx = -1;
#pragma unroll
    for (int k = 0; k <= NBINS; ++k) idx += (xi >= cw[k]) ? 1 : 0;
    idx = min(max(idx, 0), NBINS - 1);

    float in_cw = 0.f, in_w = 1.f, in_ch = 0.f, in_h = 1.f, dr0 = 0.f, dr1 = 0.f;
#pragma unroll
    for (int k = 0; k < NBINS; ++k)
      if (k == idx) {
        in_cw = cw[k]; in_w = cw[k + 1] - cw[k];
        in_ch = ch[k]; in_h = ch[k + 1] - ch[k];
        dr0 = ddraw[k]; dr1 = ddraw[k + 1];
      }

    const float in_d  = 0.001f + softplus_f(dr0);
    const float d_p   = 0.001f + softplus_f(dr1);
    const float delta = in_h / in_w;
    const float th    = (xi - in_cw) / in_w;
    const float th1   = th * (1.0f - th);
    const float num   = in_h * (delta * th * th + in_d * th1);
    const float den   = delta + (in_d + d_p - 2.0f * delta) * th1;
    const float omt   = 1.0f - th;
    const float dnum  = delta * delta * (d_p * th * th + 2.0f * delta * th1 + in_d * omt * omt);

    const float yv  = ins ? (in_ch + num / den) : x1v;
    const float lad = ins ? (logf(dnum) - 2.0f * logf(den)) : 0.0f;

    out[(b * 2 + 0) * TT + gt] = xsh[j] * m;
    out[(b * 2 + 1) * TT + gt] = yv * m;
    lsum = lad * m;
  }
  __syncthreads();   // protect xsh reuse below

  // ---- deterministic per-tile logdet partial ----
  float v = lsum;
#pragma unroll
  for (int o = 16; o > 0; o >>= 1) v += __shfl_down(v, o, 32);
  if (lane == 0) xsh[wave] = v;
  __syncthreads();
  if (tid == 0) {
    float s = 0.f;
#pragma unroll
    for (int w = 0; w < 8; ++w) s += xsh[w];
    partials[b * gridDim.x + tile] = s;
  }
}

// deterministic reduction of per-tile partials -> logdet[b]
__global__ void logdet_reduce(const float* __restrict__ partials,
                              float* __restrict__ out, int ntiles)
{
  __shared__ float sh[64];
  const int b = blockIdx.x;
  const int t = threadIdx.x;
  sh[t] = partials[b * ntiles + t];
  __syncthreads();
  for (int s = 32; s > 0; s >>= 1) {
    if (t < s) sh[t] += sh[t + s];
    __syncthreads();
  }
  if (t == 0) out[BB * 2 * TT + b] = sh[0];
}

extern "C" void kernel_launch(void* const* d_in, const int* in_sizes, int n_in,
                              void* d_out, int out_size, void* d_ws, size_t ws_size,
                              hipStream_t stream)
{
  const float* x      = (const float*)d_in[0];
  const float* xmask  = (const float*)d_in[1];
  const float* w_pre  = (const float*)d_in[2];
  const float* b_pre  = (const float*)d_in[3];
  const float* sep_w  = (const float*)d_in[4];
  const float* sep_b  = (const float*)d_in[5];
  const float* pw_w   = (const float*)d_in[6];
  const float* pw_b   = (const float*)d_in[7];
  const float* n1_g   = (const float*)d_in[8];
  const float* n1_b   = (const float*)d_in[9];
  const float* n2_g   = (const float*)d_in[10];
  const float* n2_b   = (const float*)d_in[11];
  const float* w_proj = (const float*)d_in[12];
  const float* b_proj = (const float*)d_in[13];
  float* out      = (float*)d_out;
  float* partials = (float*)d_ws;   // 16*64 floats

  // 2 * 192*168*4 (h,z f32) + 168*200*2 (y bf16) + 2*168*4 (x0,mask) = 326,592 B (< 320 KiB WGP LDS)
  const size_t smem = (size_t)(CC * W0P * 4) * 2 + (size_t)W0P * LDY * 2 + (size_t)W0P * 4 * 2;
  hipFuncSetAttribute((const void*)convflow_kernel,
                      hipFuncAttributeMaxDynamicSharedMemorySize, (int)smem);

  dim3 grid(TT / TILE, BB);   // 64 tiles x 16 batches
  convflow_kernel<<<grid, 256, smem, stream>>>(
      x, xmask, w_pre, b_pre, sep_w, sep_b, pw_w, pw_b,
      n1_g, n1_b, n2_g, n2_b, w_proj, b_proj, out, partials);

  logdet_reduce<<<dim3(BB), 64, 0, stream>>>(partials, out, TT / TILE);
}
